// ScannedMultiLayerGRU_65438121722403
// MI455X (gfx1250) — matrix operand order repaired and checked by reading, not compiled, not run on hardware
//
#include <hip/hip_runtime.h>
#include <math.h>

// Problem constants (from reference): L=3, T=512, B=64, W=1024
#define T_STEPS 512
#define BATCH   64
#define WID     1024
#define GW      3072            // 3*W gate width
#define NLAYER  3
#define KBLK    32              // 1024 / 32 k-blocks per GEMM
#define NTG     192             // 3072 / 16 column tiles in gate space
#define NBLOCKS 32              // persistent grid: 32 WGs x 256 thr (co-resident)
#define STATE_ELEMS (NLAYER * BATCH * WID)                 // 196608
#define XSTEP   ((size_t)BATCH * WID)                      // 65536 elems / matrix
#define XBYTES  (XSTEP * 2)                                // 131072 B packed-x/step
#define PACK_ELEMS  ((size_t)NLAYER * 2 * KBLK * NTG * 32 * 16)  // 18,874,368 bf16

typedef __attribute__((ext_vector_type(16))) __bf16   v16bf;
typedef __attribute__((ext_vector_type(8)))  __bf16   v8bf;
typedef __attribute__((ext_vector_type(8)))  float    v8f;
typedef __attribute__((ext_vector_type(8)))  unsigned v8u;
typedef __attribute__((ext_vector_type(4)))  unsigned u32x4;
typedef __attribute__((ext_vector_type(8)))  int      i32x8;
typedef __attribute__((ext_vector_type(4)))  int      i32x4;

// ---------------------------------------------------------------------------
// ws layout, ONE base pointer so every access keeps addrspace(1):
//   [PW_OFF ] packed weights, bf16 B-fragment order            (37.75 MB)
//   [PX_OFF ] packed ins, bf16 A-fragment order, per step      (64 MB)
//   [PH_OFF ] packed h state, A-fragment order, 2 buffers      (0.75 MB)
//   [F32_OFF] h state fp32, 2 buffers (z*h + final carry)      (1.5 MB)
//   [BAR_OFF] grid barrier words
// ---------------------------------------------------------------------------
constexpr size_t PW_OFF  = 0;                                  // bf16 elems
constexpr size_t PX_OFF  = PACK_ELEMS;
constexpr size_t PH_OFF  = PX_OFF + (size_t)T_STEPS * XSTEP;
constexpr size_t PH_BUF  = (size_t)NLAYER * XSTEP;             // 196608
constexpr size_t F32_OFF = (PH_OFF + 2 * PH_BUF) / 2;          // float elems
constexpr size_t BAR_OFF = F32_OFF + 2 * (size_t)STATE_ELEMS;  // float elems

// A-matrix (16x32 MxK) per-lane k map (ISA 7.12.2).
__device__ __forceinline__ int amap_k(int lane, int e) {
  return (lane >> 4) * 8 + ((e < 8) ? e : 16 + (e - 8));
}

// ---------------------------------------------------------------------------
// One-time pre-packs
// ---------------------------------------------------------------------------
__global__ __launch_bounds__(256) void gru_pack_weights(
    const float* __restrict__ Wi, const float* __restrict__ Wh,
    __bf16* __restrict__ ws) {
  size_t id = (size_t)blockIdx.x * 256 + threadIdx.x;
  if (id >= PACK_ELEMS) return;
  int e     = (int)(id & 15);
  int lane  = (int)((id >> 4) & 31);
  size_t q  = id >> 9;
  int nt    = (int)(q % NTG);  q /= NTG;
  int kb    = (int)(q % KBLK); q /= KBLK;
  int which = (int)(q & 1);    q >>= 1;
  int layer = (int)q;
  int k = kb * 32 + amap_k(lane, e);        // B 32x16: same k map as A fragment
  int n = nt * 16 + (lane & 15);
  const float* W = which ? Wh : Wi;
  ws[PW_OFF + id] = (__bf16)W[((size_t)layer * WID + k) * GW + n];
}

__global__ __launch_bounds__(256) void gru_pack_x(
    const float* __restrict__ ins, __bf16* __restrict__ ws) {
  size_t id = (size_t)blockIdx.x * 256 + threadIdx.x;
  if (id >= (size_t)T_STEPS * XSTEP) return;
  int e = (int)(id & 15), lane = (int)((id >> 4) & 31);
  int kb = (int)((id >> 9) & 31), mtile = (int)((id >> 14) & 3);
  size_t t = id >> 16;
  int row = mtile * 16 + (lane & 15);
  int k = kb * 32 + amap_k(lane, e);
  ws[PX_OFF + id] = (__bf16)ins[(t * BATCH + row) * WID + k];
}

__global__ __launch_bounds__(256) void gru_init(
    const float* __restrict__ carry0, __bf16* __restrict__ ws) {
  size_t i = (size_t)blockIdx.x * 256 + threadIdx.x;
  float* fbase = (float*)ws;
  unsigned* bar = (unsigned*)(fbase + BAR_OFF);
  if (i < 4) bar[i] = 0u;
  if (i < (size_t)STATE_ELEMS) {
    fbase[F32_OFF + i] = carry0[i];
    int e = (int)(i & 15), lane = (int)((i >> 4) & 31);
    int kb = (int)((i >> 9) & 31), mtile = (int)((i >> 14) & 3);
    int l = (int)(i >> 16);
    int row = mtile * 16 + (lane & 15);
    int k = kb * 32 + amap_k(lane, e);
    ws[PH_OFF + i] = (__bf16)carry0[(size_t)l * XSTEP + (size_t)row * WID + k];
  }
}

// ---------------------------------------------------------------------------
// Fast gate math (v_rcp_f32, correct +-inf saturation).
// ---------------------------------------------------------------------------
__device__ __forceinline__ float fsigmoid(float x) {
  return __builtin_amdgcn_rcpf(1.0f + __expf(-x));
}
__device__ __forceinline__ float ftanh(float x) {
  float t = __expf(2.0f * x);
  return 1.0f - 2.0f * __builtin_amdgcn_rcpf(t + 1.0f);
}

// Device-scope sense-reversal barrier (32 co-resident WGs -> deadlock-free).
__device__ __forceinline__ void grid_barrier(unsigned* cnt, unsigned* gen,
                                             unsigned nb) {
  __syncthreads();
  if (threadIdx.x == 0) {
    __threadfence();
    unsigned g = __hip_atomic_load(gen, __ATOMIC_RELAXED, __HIP_MEMORY_SCOPE_AGENT);
    unsigned a = __hip_atomic_fetch_add(cnt, 1u, __ATOMIC_ACQ_REL, __HIP_MEMORY_SCOPE_AGENT);
    if (a == nb - 1u) {
      __hip_atomic_store(cnt, 0u, __ATOMIC_RELAXED, __HIP_MEMORY_SCOPE_AGENT);
      __hip_atomic_fetch_add(gen, 1u, __ATOMIC_RELEASE, __HIP_MEMORY_SCOPE_AGENT);
    } else {
      while (__hip_atomic_load(gen, __ATOMIC_ACQUIRE, __HIP_MEMORY_SCOPE_AGENT) == g) {
        __builtin_amdgcn_s_sleep(2);
      }
    }
    __threadfence();
  }
  __syncthreads();
}

// ---------------------------------------------------------------------------
// TDM: stage one step's packed-x (131072 B) global -> LDS. D# per ISA ch.8:
// one row of 16384 8-byte elements (2D degenerate). Issued by wave 0 only
// (EXEC is ignored by tensor ops); completion via s_wait_tensorcnt.
// ---------------------------------------------------------------------------
#if __has_builtin(__builtin_amdgcn_tensor_load_to_lds)
#define HAVE_TDM 1
__device__ __forceinline__ void tdm_stage_x(const __bf16* src, unsigned lds_off) {
  unsigned long long ga = (unsigned long long)(const void*)src;  // AS1: VA
  u32x4 g0;
  g0.x = 1u;                                          // count=1, user D#
  g0.y = lds_off;                                     // LDS byte address
  g0.z = (unsigned)(ga & 0xFFFFFFFFull);              // global_addr[31:0]
  g0.w = (unsigned)((ga >> 32) & 0x1FFFFFFull)        // global_addr[56:32]
         | (2u << 30);                                // type=2 ("image")
  i32x8 g1 = {};
  g1[0] = (3 << 16);                                  // data_size=3 (8B), mask=0
  g1[1] = (int)(0x4000u << 16);                       // tensor_dim0 = 16384 (lo16)
  g1[2] = (1 << 16);                                  // dim0 hi=0 ; tensor_dim1 = 1
  g1[3] = (int)(0x4000u << 16);                       // dim1 hi=0 ; tile_dim0 = 16384
  g1[4] = 0;                                          // tile_dim1/2 unused
  g1[5] = 0x4000;                                     // tensor_dim0_stride = 16384
  g1[6] = 0;
  g1[7] = 0;
  i32x4 gz = {};
#if defined(__clang_major__) && (__clang_major__ >= 23)
  i32x8 gz8 = {};
  __builtin_amdgcn_tensor_load_to_lds(g0, g1, gz, gz, gz8, 0);
#else
  __builtin_amdgcn_tensor_load_to_lds(g0, g1, gz, gz, 0);
#endif
}
#else
#define HAVE_TDM 0
#endif

// ---------------------------------------------------------------------------
// k-loop: 32 iters x (2 A-frags + 6 B-frags + 6 WMMA). forceinline so each
// call site gets its own address-space inference (LDS x vs global x).
// ---------------------------------------------------------------------------
__device__ __forceinline__ void kloop(
    const __bf16* xp, const __bf16* hp,
    const __bf16* wi_p, const __bf16* wh_p, unsigned hmw,
    v8f& ai0, v8f& ai1, v8f& ai2, v8f& ah0, v8f& ah1, v8f& ah2) {
  const size_t kstride = (size_t)NTG * 512;
  const size_t gstride = (size_t)64 * 512;
  #pragma unroll 2
  for (int kb = 0; kb < KBLK; ++kb) {
    v16bf axf = *(const v16bf*)xp;
    v8u   hrw = *(const v8u*)hp;
    v16bf ahf = __builtin_bit_cast(v16bf, hrw & hmw);   // episode reset
    v16bf w0 = *(const v16bf*)(wi_p);
    v16bf w1 = *(const v16bf*)(wi_p + gstride);
    v16bf w2 = *(const v16bf*)(wi_p + 2 * gstride);
    v16bf u0 = *(const v16bf*)(wh_p);
    v16bf u1 = *(const v16bf*)(wh_p + gstride);
    v16bf u2 = *(const v16bf*)(wh_p + 2 * gstride);
    __builtin_prefetch(wi_p + kstride, 0, 1);           // L2-hot weights
    __builtin_prefetch(wh_p + kstride, 0, 1);
    ai0 = __builtin_amdgcn_wmma_f32_16x16x32_bf16(false, axf, false, w0,
                                                  (short)0, ai0, false, false);
    ai1 = __builtin_amdgcn_wmma_f32_16x16x32_bf16(false, axf, false, w1,
                                                  (short)0, ai1, false, false);
    ai2 = __builtin_amdgcn_wmma_f32_16x16x32_bf16(false, axf, false, w2,
                                                  (short)0, ai2, false, false);
    ah0 = __builtin_amdgcn_wmma_f32_16x16x32_bf16(false, ahf, false, u0,
                                                  (short)0, ah0, false, false);
    ah1 = __builtin_amdgcn_wmma_f32_16x16x32_bf16(false, ahf, false, u1,
                                                  (short)0, ah1, false, false);
    ah2 = __builtin_amdgcn_wmma_f32_16x16x32_bf16(false, ahf, false, u2,
                                                  (short)0, ah2, false, false);
    xp += 512;  hp += 512;
    wi_p += kstride;  wh_p += kstride;
  }
}

// ---------------------------------------------------------------------------
// Persistent GRU scan. 32 blocks x 256 threads (8 wave32). Block b owns
// hidden cols [b*32, b*32+32); wave w: mtile=w&3, ntile=w>>2. Layer-0 A
// operand comes from LDS (TDM-prefetched one step ahead, hidden behind
// layers 1-2 of the previous step); layers 1-2 read packed-A h from global.
// ---------------------------------------------------------------------------
__global__ __launch_bounds__(256, 1) void gru_scan(
    const unsigned char* __restrict__ resets, const float* __restrict__ bh,
    __bf16* __restrict__ ws, float* __restrict__ out) {
  const int lane  = threadIdx.x & 31;
  const int wave  = threadIdx.x >> 5;
  const int mtile = wave & 3;
  const int m0    = mtile * 16;
  const int chid0 = blockIdx.x * 32 + (wave >> 2) * 16;
  const int arow  = m0 + (lane & 15);
  const int khalf = lane >> 4;
  const int cr    = chid0 + (lane & 15);
  const size_t ntg0    = (size_t)(chid0 >> 4);
  const size_t kstride = (size_t)NTG * 512;
  const size_t abase   = ((size_t)mtile * KBLK * 32 + lane) * 16;

  float* fbase  = (float*)ws;
  unsigned* bar = (unsigned*)(fbase + BAR_OFF);
  unsigned* cnt = bar;
  unsigned* gen = bar + 1;
  float* ys = out + STATE_ELEMS;

  // [0, 128KB): staged packed-x ; [128KB, 136KB): 16x16 transpose tiles
  __shared__ __attribute__((aligned(128))) unsigned char ldsAll[XBYTES + 8 * 256 * 4];
  __bf16* ldsX = (__bf16*)ldsAll;
  float*  ldsT = (float*)(ldsAll + XBYTES);
  float*  myt  = ldsT + wave * 256;
  const unsigned ldsx_off = (unsigned)(size_t)ldsX;  // low 32b of flat = LDS offset

  // Stage packed-x for t=0.
#if HAVE_TDM
  if (wave == 0) tdm_stage_x(ws + PX_OFF, ldsx_off);
#else
  for (int i = threadIdx.x; i < (int)(XBYTES / 16); i += 256)
    ((float4*)ldsX)[i] = ((const float4*)(ws + PX_OFF))[i];
#endif

  for (int t = 0; t < T_STEPS; ++t) {
#if HAVE_TDM
    if (wave == 0) __builtin_amdgcn_s_wait_tensorcnt(0);
#endif
    __syncthreads();                                   // ldsX ready for layer 0

    const size_t f32old = F32_OFF + (size_t)(t & 1) * STATE_ELEMS;
    const size_t f32new = F32_OFF + (size_t)((t + 1) & 1) * STATE_ELEMS;
    const size_t phold  = PH_OFF + (size_t)(t & 1) * PH_BUF;
    const size_t phnew  = PH_OFF + (size_t)((t + 1) & 1) * PH_BUF;
    const unsigned char* rst = resets + t * BATCH;     // numpy bool_: 1 byte
    const unsigned hmw = rst[arow] ? 0u : 0xFFFFFFFFu;

    for (int l = 0; l < NLAYER; ++l) {
      const __bf16* hp   = ws + phold + (size_t)l * XSTEP + abase;
      const __bf16* wi_p = ws + PW_OFF + (size_t)(l * 2 + 0) * KBLK * kstride
                              + ntg0 * 512 + (size_t)lane * 16;
      const __bf16* wh_p = ws + PW_OFF + (size_t)(l * 2 + 1) * KBLK * kstride
                              + ntg0 * 512 + (size_t)lane * 16;

      v8f ai0 = {}, ai1 = {}, ai2 = {};                // x @ Wi (r|z|n)
      v8f ah0 = {}, ah1 = {}, ah2 = {};                // h @ Wh (r|z|n)

      if (l == 0) {
        kloop(ldsX + abase, hp, wi_p, wh_p, hmw, ai0, ai1, ai2, ah0, ah1, ah2);
      } else {
        kloop(ws + phnew + (size_t)(l - 1) * XSTEP + abase, hp, wi_p, wh_p, hmw,
              ai0, ai1, ai2, ah0, ah1, ah2);
      }

      // Gate math; C/D layout: lane col = lane&15, VGPR v -> row m0+v+8*khalf.
      const float bhr = bh[l * GW + cr];
      const float bhz = bh[l * GW + WID + cr];
      const float bhn = bh[l * GW + 2 * WID + cr];
      #pragma unroll
      for (int v = 0; v < 8; ++v) {
        const int m = m0 + v + khalf * 8;
        const float r = fsigmoid(ai0[v] + ah0[v] + bhr);
        const float z = fsigmoid(ai1[v] + ah1[v] + bhz);
        const float n = ftanh(ai2[v] + r * (ah2[v] + bhn));
        const float hm = rst[m] ? 0.0f : 1.0f;
        const float ho = hm * fbase[f32old + (size_t)l * XSTEP + (size_t)m * WID + cr];
        const float hn = (1.0f - z) * n + z * ho;
        fbase[f32new + (size_t)l * XSTEP + (size_t)m * WID + cr] = hn;
        if (l == NLAYER - 1) ys[((size_t)t * BATCH + m) * WID + cr] = hn;
        myt[(v + 8 * khalf) * 16 + (lane & 15)] = hn;  // stage for transpose
      }
      __syncthreads();  // transpose tiles complete; ldsX reads of step t done

      if (l == 0 && t + 1 < T_STEPS) {
        // Prefetch next step's packed-x into LDS; hidden behind layers 1-2.
#if HAVE_TDM
        if (wave == 0) tdm_stage_x(ws + PX_OFF + (size_t)(t + 1) * XSTEP, ldsx_off);
#else
        for (int i = threadIdx.x; i < (int)(XBYTES / 16); i += 256)
          ((float4*)ldsX)[i] =
              ((const float4*)(ws + PX_OFF + (size_t)(t + 1) * XSTEP))[i];
#endif
      }

      // Read back in A-fragment order, cvt to bf16, store packed-A h.
      {
        const float* rb = myt + (lane & 15) * 16 + khalf * 8;
        float4 c0 = *(const float4*)(rb);
        float4 c1 = *(const float4*)(rb + 4);
        v8bf o;
        o[0] = (__bf16)c0.x; o[1] = (__bf16)c0.y;
        o[2] = (__bf16)c0.z; o[3] = (__bf16)c0.w;
        o[4] = (__bf16)c1.x; o[5] = (__bf16)c1.y;
        o[6] = (__bf16)c1.z; o[7] = (__bf16)c1.w;
        const size_t kbt = (size_t)(chid0 >> 5);
        const int es = (chid0 & 16) ? 8 : 0;
        *(v8bf*)(ws + phnew + (size_t)l * XSTEP
                 + ((size_t)mtile * KBLK + kbt) * 512 + (size_t)lane * 16 + es) = o;
      }
      grid_barrier(cnt, gen, NBLOCKS);  // next layer/step consumes h_new
    }
  }

  // T even -> final state sits in buffer 0. Copy to d_out[0:STATE_ELEMS).
  for (size_t i = (size_t)blockIdx.x * 256 + threadIdx.x; i < (size_t)STATE_ELEMS;
       i += (size_t)NBLOCKS * 256)
    out[i] = fbase[F32_OFF + i];
}

// ---------------------------------------------------------------------------
// Launch. Inputs: ins f32[T,B,W], resets bool[T,B], carry0 f32[L,B,W],
// Wi f32[L,W,3W], Wh f32[L,W,3W], bh f32[L,3W].
// d_out = final_carry (L*B*W f32) ++ ys (T*B*W f32).
// ---------------------------------------------------------------------------
extern "C" void kernel_launch(void* const* d_in, const int* in_sizes, int n_in,
                              void* d_out, int out_size, void* d_ws, size_t ws_size,
                              hipStream_t stream) {
  const float*         ins    = (const float*)d_in[0];
  const unsigned char* resets = (const unsigned char*)d_in[1];
  const float*         carry0 = (const float*)d_in[2];
  const float*         Wi     = (const float*)d_in[3];
  const float*         Wh     = (const float*)d_in[4];
  const float*         bh     = (const float*)d_in[5];
  float*  out = (float*)d_out;
  __bf16* ws  = (__bf16*)d_ws;

  gru_init<<<(STATE_ELEMS + 255) / 256, 256, 0, stream>>>(carry0, ws);
  gru_pack_weights<<<(unsigned)((PACK_ELEMS + 255) / 256), 256, 0, stream>>>(Wi, Wh, ws);
  gru_pack_x<<<(unsigned)(((size_t)T_STEPS * XSTEP + 255) / 256), 256, 0, stream>>>(ins, ws);
  gru_scan<<<NBLOCKS, 256, 0, stream>>>(resets, bh, ws, out);
}
